// GraphAttentionNetwork_1408749273476
// MI455X (gfx1250) — compile-verified, matrix operand order checked by reading, not run on hardware
//
#include <hip/hip_runtime.h>

typedef __attribute__((ext_vector_type(16))) __bf16 v16bf;
typedef __attribute__((ext_vector_type(8)))  __bf16 v8bf;
typedef __attribute__((ext_vector_type(8)))  float  v8f;

#define GAT_C 64   // per-head channels everywhere in this model

// ---------- helpers ----------
__device__ __forceinline__ __bf16 f2bf(float f) {
  // round-to-nearest-even f32 -> bf16 (used only in memory-bound side passes)
  unsigned u = __float_as_uint(f);
  unsigned r = (u + 0x7FFFu + ((u >> 16) & 1u)) >> 16;
  unsigned short s = (unsigned short)r;
  return __builtin_bit_cast(__bf16, s);
}
// order-preserving float <-> uint (atomic float max incl. negatives)
__device__ __forceinline__ unsigned f2ord(float f) {
  unsigned u = __float_as_uint(f);
  return (u & 0x80000000u) ? ~u : (u | 0x80000000u);
}
__device__ __forceinline__ float ord2f(unsigned u) {
  return __uint_as_float((u & 0x80000000u) ? (u & 0x7FFFFFFFu) : ~u);
}

// ---------- side passes: f32 -> bf16 activation copy, f32 -> bf16^T weights ----------
__global__ void f32_to_bf16_kernel(const float* __restrict__ in, __bf16* __restrict__ out,
                                   long long n) {
  long long i = (blockIdx.x * (long long)blockDim.x + threadIdx.x) * 4;
  if (i + 3 < n) {
    float4 v = *(const float4*)(in + i);
    out[i + 0] = f2bf(v.x); out[i + 1] = f2bf(v.y);
    out[i + 2] = f2bf(v.z); out[i + 3] = f2bf(v.w);
  } else {
    for (int j = 0; j < 4; ++j) if (i + j < n) out[i + j] = f2bf(in[i + j]);
  }
}

__global__ void w_transpose_bf16_kernel(const float* __restrict__ W, __bf16* __restrict__ WT,
                                        int K, int Nout) {
  int i = blockIdx.x * blockDim.x + threadIdx.x;   // i = n*K + k
  if (i >= K * Nout) return;
  int nn = i / K, k = i - nn * K;
  WT[i] = f2bf(W[(size_t)k * Nout + nn]);
}

// ---------- WMMA GEMM:  C[M,Nout] = act(A[M,K] @ W[K,Nout] (+ bias)) ----------
// A: bf16 row-major [M,K].  WT: bf16 [Nout,K] (W transposed).
// One wave computes a 16 x (16*NT) tile of C, reusing one A fragment across NT
// wmma issues per K-step.  Per the ISA 16-bit fragment layout
// (K = 2*(r&3) + 16*(r>>2) + 8*laneHi + p), each lane's fragment is two
// contiguous 8-element (16-byte) runs -> two global_load_b128 per fragment.
template <int NT>
__global__ __launch_bounds__(256)
void gemm_wmma_bf16_kernel(const __bf16* __restrict__ A, const __bf16* __restrict__ WT,
                           const float* __restrict__ bias, float* __restrict__ C,
                           int M, int K, int Nout, int doRelu)
{
  const int lane    = threadIdx.x & 31;
  const int waveId  = blockIdx.x * (blockDim.x >> 5) + (threadIdx.x >> 5);
  const int groupsN = Nout / (16 * NT);
  const int waves   = (M >> 4) * groupsN;
  if (waveId >= waves) return;                 // uniform per wave: EXEC stays all-1s
  const int tm  = waveId / groupsN;
  const int tg  = waveId - tm * groupsN;
  const int l15 = lane & 15;
  const int hi  = lane >> 4;
  const int row = (tm << 4) + l15;
  const int colBase = tg * (16 * NT);

  const __bf16* arow = A + (size_t)row * K;
  v8f acc[NT];
#pragma unroll
  for (int t = 0; t < NT; ++t) acc[t] = (v8f){};

  for (int k0 = 0; k0 < K; k0 += 32) {
    const v8bf a0 = *(const v8bf*)(arow + k0 + (hi << 3));
    const v8bf a1 = *(const v8bf*)(arow + k0 + 16 + (hi << 3));
    const v16bf a = __builtin_shufflevector(a0, a1, 0,1,2,3,4,5,6,7,8,9,10,11,12,13,14,15);
#pragma unroll
    for (int t = 0; t < NT; ++t) {
      const __bf16* wcol = WT + (size_t)(colBase + t * 16 + l15) * K;
      const v8bf b0 = *(const v8bf*)(wcol + k0 + (hi << 3));
      const v8bf b1 = *(const v8bf*)(wcol + k0 + 16 + (hi << 3));
      const v16bf b = __builtin_shufflevector(b0, b1, 0,1,2,3,4,5,6,7,8,9,10,11,12,13,14,15);
      acc[t] = __builtin_amdgcn_wmma_f32_16x16x32_bf16(false, a, false, b,
                                                       (short)0, acc[t], false, false);
    }
  }
#pragma unroll
  for (int t = 0; t < NT; ++t) {
    const int col = colBase + t * 16 + l15;
    const float bb = bias ? bias[col] : 0.0f;
#pragma unroll
    for (int r = 0; r < 8; ++r) {              // C/D layout: VGPR r -> M = r + 8*hi
      float v = acc[t][r] + bb;
      if (doRelu) v = fmaxf(v, 0.0f);
      C[(size_t)((tm << 4) + (hi << 3) + r) * Nout + col] = v;
    }
  }
}

// ---------- per-node attention logits: aS[n,h] = <h[n,h,:], a_src[h,:]> ----------
__global__ void node_alpha_kernel(const float* __restrict__ hf,
                                  const float* __restrict__ a_src,
                                  const float* __restrict__ a_dst,
                                  float* __restrict__ aS, float* __restrict__ aD,
                                  int N, int heads)
{
  int i = blockIdx.x * blockDim.x + threadIdx.x;   // i = n*heads + h
  if (i >= N * heads) return;
  int h = i % heads;
  const float* hp  = hf + (size_t)i * GAT_C;
  const float* asp = a_src + h * GAT_C;
  const float* adp = a_dst + h * GAT_C;
  float ss = 0.f, dd = 0.f;
  for (int c = 0; c < GAT_C; c += 4) {
    float4 v = *(const float4*)(hp + c);
    float4 s4 = *(const float4*)(asp + c);
    float4 d4 = *(const float4*)(adp + c);
    ss += v.x * s4.x + v.y * s4.y + v.z * s4.z + v.w * s4.w;
    dd += v.x * d4.x + v.y * d4.y + v.z * d4.z + v.w * d4.w;
  }
  aS[i] = ss; aD[i] = dd;
}

__global__ void fill_u32_kernel(unsigned* __restrict__ p, unsigned v, long long n) {
  long long i = blockIdx.x * (long long)blockDim.x + threadIdx.x;
  if (i < n) p[i] = v;
}

__device__ __forceinline__ void edge_sd(const int* ei, int E, int e, int& s, int& d) {
  if (e < E) { s = ei[e]; d = ei[E + e]; } else { s = d = e - E; }  // self-loops appended
}

__device__ __forceinline__ float edge_logit(const float* aS, const float* aD,
                                            int s, int d, int heads, int h) {
  float f = aS[s * heads + h] + aD[d * heads + h];
  return (f > 0.f) ? f : 0.2f * f;             // leaky_relu(0.2)
}

// pass 1: segment max of leaky-relu logits (ordered-uint atomic max)
__global__ void edge_max_kernel(const int* __restrict__ ei, int E, int N, int heads,
                                const float* __restrict__ aS, const float* __restrict__ aD,
                                unsigned* __restrict__ segM)
{
  int i = blockIdx.x * blockDim.x + threadIdx.x;
  if (i >= (E + N) * heads) return;
  int h = i % heads, e = i / heads, s, d;
  edge_sd(ei, E, e, s, d);
  float f = edge_logit(aS, aD, s, d, heads, h);
  atomicMax(&segM[d * heads + h], f2ord(f));
}

// pass 2: segment sum of exp(logit - max)
__global__ void edge_sum_kernel(const int* __restrict__ ei, int E, int N, int heads,
                                const float* __restrict__ aS, const float* __restrict__ aD,
                                const unsigned* __restrict__ segM, float* __restrict__ segS)
{
  int i = blockIdx.x * blockDim.x + threadIdx.x;
  if (i >= (E + N) * heads) return;
  int h = i % heads, e = i / heads, s, d;
  edge_sd(ei, E, e, s, d);
  float f = edge_logit(aS, aD, s, d, heads, h);
  int di = d * heads + h;
  atomicAdd(&segS[di], __expf(f - ord2f(segM[di])));
}

// pass 3: out[dst,h,:] += alpha * h[src,h,:]
// 16 threads per (edge,head); each handles 4 contiguous channels (float4 load,
// 4 atomic f32 adds) -> coalesced 256B per (edge,head).
__global__ void edge_aggr_kernel(const int* __restrict__ ei, int E, int N, int heads,
                                 const float* __restrict__ aS, const float* __restrict__ aD,
                                 const unsigned* __restrict__ segM, const float* __restrict__ segS,
                                 const float* __restrict__ hf, float* __restrict__ out)
{
  long long i = blockIdx.x * (long long)blockDim.x + threadIdx.x;
  long long total = (long long)(E + N) * heads * 16;
  if (i >= total) return;
  int c0 = (int)(i & 15) * 4;                  // channel group
  long long eh = i >> 4;
  int h = (int)(eh % heads);
  int e = (int)(eh / heads);
  int s, d;
  edge_sd(ei, E, e, s, d);
  float f = edge_logit(aS, aD, s, d, heads, h);
  int di = d * heads + h;
  float alpha = __expf(f - ord2f(segM[di])) / (segS[di] + 1e-16f);
  const float4 hv = *(const float4*)(hf + (size_t)(s * heads + h) * GAT_C + c0);
  float* od = out + (size_t)di * GAT_C + c0;
  atomicAdd(&od[0], hv.x * alpha);
  atomicAdd(&od[1], hv.y * alpha);
  atomicAdd(&od[2], hv.z * alpha);
  atomicAdd(&od[3], hv.w * alpha);
}

__global__ void bias_act_kernel(float* __restrict__ buf, const float* __restrict__ bias,
                                long long n, int dim, int doRelu)
{
  long long i = blockIdx.x * (long long)blockDim.x + threadIdx.x;
  if (i >= n) return;
  float v = buf[i] + bias[(int)(i % dim)];
  if (doRelu) v = fmaxf(v, 0.0f);
  buf[i] = v;
}

__global__ void final_out_kernel(const float* __restrict__ h32, const float* __restrict__ Wo2,
                                 const float* __restrict__ bo2, float* __restrict__ out, int N)
{
  int n = blockIdx.x * blockDim.x + threadIdx.x;
  if (n >= N) return;
  float sum = bo2[0];
  for (int c = 0; c < 32; c += 4) {
    float4 v = *(const float4*)(h32 + (size_t)n * 32 + c);
    float4 w = *(const float4*)(Wo2 + c);
    sum += v.x * w.x + v.y * w.y + v.z * w.z + v.w * w.w;
  }
  out[n] = sum;
}

// ---------- host orchestration ----------
extern "C" void kernel_launch(void* const* d_in, const int* in_sizes, int n_in,
                              void* d_out, int out_size, void* d_ws, size_t ws_size,
                              hipStream_t stream)
{
  const float* x     = (const float*)d_in[0];
  const float* W_emb = (const float*)d_in[1];
  const float* b_emb = (const float*)d_in[2];
  const float* W0    = (const float*)d_in[3];
  const float* as0   = (const float*)d_in[4];
  const float* ad0   = (const float*)d_in[5];
  const float* b0    = (const float*)d_in[6];
  const float* W1    = (const float*)d_in[7];
  const float* as1   = (const float*)d_in[8];
  const float* ad1   = (const float*)d_in[9];
  const float* b1    = (const float*)d_in[10];
  const float* W2    = (const float*)d_in[11];
  const float* as2   = (const float*)d_in[12];
  const float* ad2   = (const float*)d_in[13];
  const float* b2    = (const float*)d_in[14];
  const float* Wo1   = (const float*)d_in[15];
  const float* bo1   = (const float*)d_in[16];
  const float* Wo2   = (const float*)d_in[17];
  const float* bo2   = (const float*)d_in[18];
  const int*   ei    = (const int*)d_in[19];

  const int N = in_sizes[0] / 128;     // 50000 (multiple of 16)
  const int E = in_sizes[19] / 2;      // 800000

  // workspace layout
  float*    h0   = (float*)d_ws;                  // N*64 f32
  float*    hA   = h0 + (size_t)N * 64;           // N*256 f32
  float*    hB   = hA + (size_t)N * 256;          // N*256 f32
  float*    aS   = hB + (size_t)N * 256;          // N*4 f32
  float*    aD   = aS + (size_t)N * 4;            // N*4 f32
  unsigned* segM = (unsigned*)(aD + (size_t)N * 4);   // N*4 u32
  float*    segS = (float*)(segM + (size_t)N * 4);    // N*4 f32
  __bf16*   Abf  = (__bf16*)(segS + (size_t)N * 4);   // N*256 bf16
  __bf16*   WTbf = Abf + (size_t)N * 256;             // up to 256*256 bf16

  const int TB = 256;
  auto nblk = [](long long n, int tb) { return (unsigned)((n + tb - 1) / tb); };
  const unsigned ORD_NEG_INF = 0x007FFFFFu;       // f2ord(-inf)
  const int Etot = E + N;

  auto gemm = [&](const float* Ain, const float* W, const float* bias, float* Cout,
                  int M, int K, int Nout, int doRelu)
  {
    long long nA = (long long)M * K;
    f32_to_bf16_kernel<<<nblk((nA + 3) / 4, TB), TB, 0, stream>>>(Ain, Abf, nA);
    w_transpose_bf16_kernel<<<nblk((long long)K * Nout, TB), TB, 0, stream>>>(W, WTbf, K, Nout);
    if (Nout % 64 == 0) {
      int waves = (M / 16) * (Nout / 64);
      gemm_wmma_bf16_kernel<4><<<(waves + 7) / 8, 256, 0, stream>>>(Abf, WTbf, bias, Cout, M, K, Nout, doRelu);
    } else {
      int waves = (M / 16) * (Nout / 32);
      gemm_wmma_bf16_kernel<2><<<(waves + 7) / 8, 256, 0, stream>>>(Abf, WTbf, bias, Cout, M, K, Nout, doRelu);
    }
  };

  // embedding: h0 = relu(x @ W_emb + b_emb)
  gemm(x, W_emb, b_emb, h0, N, 128, 64, 1);

  auto run_gat = [&](const float* hin, int K, const float* W, const float* a_s,
                     const float* a_d, const float* bias, int heads,
                     float* hfeat, float* outBuf, int doRelu)
  {
    const int outDim = heads * GAT_C;
    gemm(hin, W, nullptr, hfeat, N, K, outDim, 0);
    node_alpha_kernel<<<nblk((long long)N * heads, TB), TB, 0, stream>>>(hfeat, a_s, a_d, aS, aD, N, heads);
    fill_u32_kernel<<<nblk((long long)N * heads, TB), TB, 0, stream>>>(segM, ORD_NEG_INF, (long long)N * heads);
    fill_u32_kernel<<<nblk((long long)N * heads, TB), TB, 0, stream>>>((unsigned*)segS, 0u, (long long)N * heads);
    fill_u32_kernel<<<nblk((long long)N * outDim, TB), TB, 0, stream>>>((unsigned*)outBuf, 0u, (long long)N * outDim);
    long long teh = (long long)Etot * heads;
    edge_max_kernel <<<nblk(teh, TB), TB, 0, stream>>>(ei, E, N, heads, aS, aD, segM);
    edge_sum_kernel <<<nblk(teh, TB), TB, 0, stream>>>(ei, E, N, heads, aS, aD, segM, segS);
    edge_aggr_kernel<<<nblk(teh * 16, TB), TB, 0, stream>>>(ei, E, N, heads, aS, aD, segM, segS, hfeat, outBuf);
    bias_act_kernel <<<nblk((long long)N * outDim, TB), TB, 0, stream>>>(outBuf, bias, (long long)N * outDim, outDim, doRelu);
  };

  // layer 0: h0(N,64) -> feat hA(N,256) -> out hB(N,256), relu
  run_gat(h0, 64,  W0, as0, ad0, b0, 4, hA, hB, 1);
  // layer 1: hB -> feat hA (hA free) -> out hB (hB consumed by the GEMM's bf16 copy first)
  run_gat(hB, 256, W1, as1, ad1, b1, 4, hA, hB, 1);
  // layer 2: hB -> feat h0 (N,64; heads=1) -> out hA (first N*64), bias b2, no relu
  //          mean over 1 head == identity
  run_gat(hB, 256, W2, as2, ad2, b2, 1, h0, hA, 0);

  // MLP head: hB[:,0:32] = relu(hA(N,64) @ Wo1 + bo1)
  gemm(hA, Wo1, bo1, hB, N, 64, 32, 1);
  // out = hB(N,32) @ Wo2 + bo2
  final_out_kernel<<<nblk(N, TB), TB, 0, stream>>>(hB, Wo2, bo2, (float*)d_out, N);
}
